// lstm_based_28406913696534
// MI455X (gfx1250) — matrix operand order repaired and checked by reading, not compiled, not run on hardware
//
#include <hip/hip_runtime.h>

// ---------------------------------------------------------------------------
// MI455X / gfx1250 implementation of a 2-layer relu-LSTM + dense + softmax.
//
// Design: recurrence is serialization-bound, not BW-bound (weights ~2.5MB bf16
// fit in L2; total HBM traffic ~200MB vs 23.3TB/s). Persistent single-
// workgroup LSTM kernel, 16 wave32 waves:
//   - h (128x256 bf16) double-buffered in LDS (CDNA5: 320KB LDS per WGP)
//   - c kept in registers (8 M-tile v8f fragments per wave, across all T)
//   - z = [x_t, h] @ [[W],[U]] + b computed with v_wmma_f32_16x16x32_bf16
//   - wave w owns h-columns [16w,16w+16) => gate columns n, n+256, n+512,
//     n+768 share the same accumulator-lane mapping => gate math is pure
//     per-lane VALU on WMMA fragments.
// Weights are repacked into exact WMMA B-fragment order so each B-frag is
// two contiguous b128 loads per lane.
// ---------------------------------------------------------------------------

typedef __attribute__((ext_vector_type(16))) __bf16 v16bf;
typedef __attribute__((ext_vector_type(8)))  float  v8f;
typedef unsigned short u16;
typedef unsigned int   u32;

enum { Bc = 128, Tc = 1024, DIN = 128, Hc = 256, DOUT = 1000 };

__device__ __forceinline__ u16 f32_bf16(float f) {
  u32 u = __float_as_uint(f);
  u += 0x7FFFu + ((u >> 16) & 1u);   // round-to-nearest-even
  return (u16)(u >> 16);
}

union Frag16 { v16bf v; uint4 q[2]; };

__device__ __forceinline__ v8f wmma_bf16(v16bf a, v16bf b, v8f c) {
  // emits v_wmma_f32_16x16x32_bf16
  return __builtin_amdgcn_wmma_f32_16x16x32_bf16(
      /*neg_a=*/false, a, /*neg_b=*/false, b,
      /*c_mod=*/(short)0, c, /*reuse_a=*/false, /*reuse_b=*/false);
}

// --------------------------- f32 -> bf16 convert ---------------------------
__global__ void cvt_bf16_kernel(const float* __restrict__ src,
                                u16* __restrict__ dst, long n) {
  long i  = (long)blockIdx.x * blockDim.x + threadIdx.x;
  long st = (long)gridDim.x * blockDim.x;
  for (; i < n; i += st) dst[i] = f32_bf16(src[i]);
}

// ------------------- repack [K,1024] f32 -> WMMA B-frag bf16 ---------------
// Block (kb, nb) covers K rows [32kb,32kb+32) x N cols [16nb,16nb+16).
// dst linear = ((kb*64 + nb)*32 + lane)*16 + e, where
//   k = 32kb + 8*(lane>=16) + (e&7) + 16*(e>=8)
//   n = 16nb + (lane & 15)
__global__ void repack_b_kernel(const float* __restrict__ src,
                                u16* __restrict__ dst, int K) {
  long n = (long)K * 1024;
  long i  = (long)blockIdx.x * blockDim.x + threadIdx.x;
  long st = (long)gridDim.x * blockDim.x;
  for (; i < n; i += st) {
    long blk = i >> 9;
    int  w   = (int)(i & 511);
    int  lane = w >> 4, e = w & 15;
    long kb = blk >> 6;
    int  nb = (int)(blk & 63);
    long k  = kb * 32 + (lane >> 4) * 8 + (e & 7) + (e >> 3) * 16;
    int  nn = nb * 16 + (lane & 15);
    dst[i] = f32_bf16(src[k * 1024 + nn]);
  }
}

// ------------------------------ LSTM layer ---------------------------------
// X: bf16 [128, T, D] (D = 128 or 256), Wr: repacked [D,1024],
// Ur: repacked [256,1024], bias: f32 [1024] (i,f,g,o),
// Hseq: bf16 [128, T, 256] or null, hlast: f32 [128,256] or null.
// Launch: <<<1, 512, 131072, stream>>>  (16 wave32 waves; 2x64KB h in LDS)
__global__ __launch_bounds__(512)
void lstm_layer_wmma(const u16* __restrict__ X, int D, int T,
                     const u16* __restrict__ Wr, const u16* __restrict__ Ur,
                     const float* __restrict__ bias,
                     u16* __restrict__ Hseq, float* __restrict__ hlast) {
  extern __shared__ u16 smem[];
  u16* hb0 = smem;            // h buffer 0: 128x256 bf16 (64KB)
  u16* hb1 = smem + Bc * Hc;  // h buffer 1

  const int tid  = threadIdx.x;
  const int wv   = tid >> 5;      // 0..15: wave owns h-cols [16wv,16wv+16)
  const int lane = tid & 31;
  const int ln   = lane & 15;
  const int hi   = lane >> 4;
  const int col  = wv * 16 + ln;  // C/D frag column for this lane

  const float b_i = bias[col];
  const float b_f = bias[256 + col];
  const float b_g = bias[512 + col];
  const float b_o = bias[768 + col];

  // h0 = 0
  for (int i = tid; i < Bc * Hc / 2; i += 512) ((u32*)smem)[i] = 0u;

  v8f vzero = {};
  v8f cacc[8];
  #pragma unroll
  for (int m = 0; m < 8; ++m) cacc[m] = vzero;
  __syncthreads();

  const int Dk = D >> 5;
  for (int t = 0; t < T; ++t) {
    const u16* hcur = (t & 1) ? hb1 : hb0;
    u16*       hnxt = (t & 1) ? hb0 : hb1;

    #pragma unroll
    for (int mt = 0; mt < 8; ++mt) {
      v8f zi = vzero, zf = vzero, zg = vzero, zo = vzero;

      // ---- x_t @ W  (A rows are batch rows; two 8-elem K runs per lane) ----
      const u16* xrow = X + ((long)(mt * 16 + ln) * T + t) * D + hi * 8;
      for (int kb = 0; kb < Dk; ++kb) {
        Frag16 a;
        const u16* pa = xrow + kb * 32;
        a.q[0] = *(const uint4*)(pa);
        a.q[1] = *(const uint4*)(pa + 16);
        const u16* pb = Wr + ((long)kb << 15) + wv * 512 + lane * 16;
        Frag16 b0, b1, b2, b3;
        b0.q[0] = *(const uint4*)(pb);              b0.q[1] = *(const uint4*)(pb + 8);
        b1.q[0] = *(const uint4*)(pb + 16 * 512);   b1.q[1] = *(const uint4*)(pb + 16 * 512 + 8);
        b2.q[0] = *(const uint4*)(pb + 32 * 512);   b2.q[1] = *(const uint4*)(pb + 32 * 512 + 8);
        b3.q[0] = *(const uint4*)(pb + 48 * 512);   b3.q[1] = *(const uint4*)(pb + 48 * 512 + 8);
        zi = wmma_bf16(a.v, b0.v, zi);
        zf = wmma_bf16(a.v, b1.v, zf);
        zg = wmma_bf16(a.v, b2.v, zg);
        zo = wmma_bf16(a.v, b3.v, zo);
      }

      // ---- h_{t-1} @ U  (A from LDS, B from L2-resident repacked U) ----
      const u16* hrow = hcur + (mt * 16 + ln) * Hc + hi * 8;
      for (int kb = 0; kb < 8; ++kb) {
        Frag16 a;
        const u16* pa = hrow + kb * 32;
        a.q[0] = *(const uint4*)(pa);
        a.q[1] = *(const uint4*)(pa + 16);
        const u16* pb = Ur + ((long)kb << 15) + wv * 512 + lane * 16;
        __builtin_prefetch(pb + (1 << 15), 0, 0);   // global_prefetch next kb
        Frag16 b0, b1, b2, b3;
        b0.q[0] = *(const uint4*)(pb);              b0.q[1] = *(const uint4*)(pb + 8);
        b1.q[0] = *(const uint4*)(pb + 16 * 512);   b1.q[1] = *(const uint4*)(pb + 16 * 512 + 8);
        b2.q[0] = *(const uint4*)(pb + 32 * 512);   b2.q[1] = *(const uint4*)(pb + 32 * 512 + 8);
        b3.q[0] = *(const uint4*)(pb + 48 * 512);   b3.q[1] = *(const uint4*)(pb + 48 * 512 + 8);
        zi = wmma_bf16(a.v, b0.v, zi);
        zf = wmma_bf16(a.v, b1.v, zf);
        zg = wmma_bf16(a.v, b2.v, zg);
        zo = wmma_bf16(a.v, b3.v, zo);
      }

      // ---- gates: pure per-lane VALU on accumulator fragments ----
      float hv[8];
      #pragma unroll
      for (int e = 0; e < 8; ++e) {
        float vi = 1.f / (1.f + __expf(-(zi[e] + b_i)));
        float vf = 1.f / (1.f + __expf(-(zf[e] + b_f)));
        float vg = zg[e] + b_g; vg = vg > 0.f ? vg : 0.f;          // relu
        float vo = 1.f / (1.f + __expf(-(zo[e] + b_o)));
        float c  = vf * cacc[mt][e] + vi * vg;
        cacc[mt][e] = c;
        float rc = c > 0.f ? c : 0.f;
        hv[e] = vo * rc;
      }

      // ---- write h_t: LDS (next step) + global Hseq / h_last ----
      #pragma unroll
      for (int e = 0; e < 8; ++e) {
        int row = mt * 16 + e + 8 * hi;       // C/D layout: row = vgpr + 8*(lane>=16)
        u16 hbv = f32_bf16(hv[e]);
        hnxt[row * Hc + col] = hbv;
        if (Hseq) Hseq[((long)row * T + t) * Hc + col] = hbv;
        if (hlast && t == T - 1) hlast[row * Hc + col] = hv[e];
      }
    }
    __syncthreads();   // hnxt complete before next step reads it
  }
}

// --------------------------- dense + softmax -------------------------------
// One block per batch row. h: f32 [128,256], Wd: f32 [256,1000], bd: [1000].
__global__ __launch_bounds__(256)
void dense_softmax_kernel(const float* __restrict__ h,
                          const float* __restrict__ Wd,
                          const float* __restrict__ bd,
                          float* __restrict__ out) {
  const int b   = blockIdx.x;
  const int tid = threadIdx.x;
  __shared__ float hs[Hc];
  __shared__ float red[256];

  hs[tid] = h[b * Hc + tid];
  __syncthreads();

  float lv[4];
  float mx = -3.4e38f;
  #pragma unroll
  for (int j = 0; j < 4; ++j) {
    int c = tid + j * 256;
    float acc = -3.4e38f;
    if (c < DOUT) {
      acc = bd[c];
      for (int k = 0; k < Hc; ++k) acc += hs[k] * Wd[k * DOUT + c];
      mx = fmaxf(mx, acc);
    }
    lv[j] = acc;
  }
  red[tid] = mx; __syncthreads();
  for (int s = 128; s > 0; s >>= 1) {
    if (tid < s) red[tid] = fmaxf(red[tid], red[tid + s]);
    __syncthreads();
  }
  float m = red[0]; __syncthreads();

  float sum = 0.f;
  #pragma unroll
  for (int j = 0; j < 4; ++j) {
    int c = tid + j * 256;
    if (c < DOUT) { lv[j] = __expf(lv[j] - m); sum += lv[j]; }
  }
  red[tid] = sum; __syncthreads();
  for (int s = 128; s > 0; s >>= 1) {
    if (tid < s) red[tid] += red[tid + s];
    __syncthreads();
  }
  float inv = 1.f / red[0];
  #pragma unroll
  for (int j = 0; j < 4; ++j) {
    int c = tid + j * 256;
    if (c < DOUT) out[b * DOUT + c] = lv[j] * inv;
  }
}

// ------------------------------- launcher ----------------------------------
extern "C" void kernel_launch(void* const* d_in, const int* in_sizes, int n_in,
                              void* d_out, int out_size, void* d_ws, size_t ws_size,
                              hipStream_t stream) {
  const float* x  = (const float*)d_in[0];
  const float* W1 = (const float*)d_in[1];
  const float* U1 = (const float*)d_in[2];
  const float* b1 = (const float*)d_in[3];
  const float* W2 = (const float*)d_in[4];
  const float* U2 = (const float*)d_in[5];
  const float* b2 = (const float*)d_in[6];
  const float* Wd = (const float*)d_in[7];
  const float* bd = (const float*)d_in[8];
  float* out = (float*)d_out;

  char* ws = (char*)d_ws;
  size_t off = 0;
  u16* xb   = (u16*)(ws + off); off += (size_t)Bc * Tc * DIN * 2;   // 32 MB
  u16* W1r  = (u16*)(ws + off); off += (size_t)DIN * 1024 * 2;      // 256 KB
  u16* U1r  = (u16*)(ws + off); off += (size_t)Hc * 1024 * 2;       // 512 KB
  u16* W2r  = (u16*)(ws + off); off += (size_t)Hc * 1024 * 2;       // 512 KB
  u16* U2r  = (u16*)(ws + off); off += (size_t)Hc * 1024 * 2;       // 512 KB
  u16* Hs1  = (u16*)(ws + off); off += (size_t)Bc * Tc * Hc * 2;    // 64 MB
  float* hl = (float*)(ws + off); off += (size_t)Bc * Hc * 4;       // 128 KB

  // bf16 conversion / WMMA-fragment repacks
  cvt_bf16_kernel<<<2048, 256, 0, stream>>>(x, xb, (long)Bc * Tc * DIN);
  repack_b_kernel<<<512, 256, 0, stream>>>(W1, W1r, DIN);
  repack_b_kernel<<<1024, 256, 0, stream>>>(U1, U1r, Hc);
  repack_b_kernel<<<1024, 256, 0, stream>>>(W2, W2r, Hc);
  repack_b_kernel<<<1024, 256, 0, stream>>>(U2, U2r, Hc);

  // persistent recurrent layers (2 x 64KB h double-buffer in LDS)
  lstm_layer_wmma<<<1, 512, 2 * Bc * Hc * 2, stream>>>(
      xb, DIN, Tc, W1r, U1r, b1, Hs1, nullptr);
  lstm_layer_wmma<<<1, 512, 2 * Bc * Hc * 2, stream>>>(
      Hs1, Hc, Tc, W2r, U2r, b2, nullptr, hl);

  // tiny head
  dense_softmax_kernel<<<Bc, 256, 0, stream>>>(hl, Wd, bd, out);
}
